// Net_86784109183545
// MI455X (gfx1250) — compile-verified, matrix operand order checked by reading, not run on hardware
//
#include <hip/hip_runtime.h>
#include <hip/hip_bf16.h>

typedef __attribute__((ext_vector_type(16))) _Float16 v16h;
typedef __attribute__((ext_vector_type(8)))  _Float16 v8h;
typedef __attribute__((ext_vector_type(8)))  float    v8f;

// Problem constants (fixed by the reference's setup_inputs)
constexpr int Bsz = 128;   // batch
constexpr int T   = 1000;  // time steps
constexpr int I   = 512;   // input features
constexpr int O   = 256;   // output neurons
constexpr int SZ  = T * Bsz * O;  // elements per output record

constexpr int MROWS = 32;  // M rows (batch entries) per block

// ---------------------------------------------------------------------------
// Kernel 0: one-time convert W [O,I] f32 -> f16 into workspace (256 KB).
// ---------------------------------------------------------------------------
__global__ void snn_convert_w_f16(const float* __restrict__ W,
                                  _Float16* __restrict__ Wh) {
  int i = blockIdx.x * 256 + threadIdx.x;
  if (i < O * I) Wh[i] = (_Float16)W[i];
}

// ---------------------------------------------------------------------------
// Kernel 1: cur[t,b,o] = sum_i spk[b,t,i] * W[o,i] + bias[o]  via WMMA.
// One block = 32 M rows (fixed t, 32 consecutive b) x full N=256.
// 256 threads = 8 waves; each wave owns a 2(M) x 2(N) grid of 16x16 tiles,
// so every B fragment is reused across two A fragments (halves W traffic).
// The whole 32x512 spike band is converted f32->f16 into LDS ONCE (32 KB,
// single barrier); the K loop is then pure ds_load_b128 + v_wmma.
// ---------------------------------------------------------------------------
__global__ __launch_bounds__(256) void snn_cur_gemm(
    const float* __restrict__ spk, const _Float16* __restrict__ Wh,
    const float* __restrict__ bias, float* __restrict__ cur) {
  __shared__ _Float16 Ah[MROWS * I];  // 32 rows x 512 K, f16 = 32 KB

  const int blk = blockIdx.x;        // T*Bsz/32 = 4000 blocks
  const int t   = blk >> 2;          // 4 blocks per timestep (Bsz/32)
  const int b0  = (blk & 3) * MROWS;

  const int tid   = threadIdx.x;
  const int lane  = tid & 31;
  const int wave  = tid >> 5;
  const int mrow  = lane & 15;
  const int khalf = (lane >> 4) << 3;  // 0 for lanes 0-15, 8 for lanes 16-31
  const int n0    = wave * 32;         // wave's first N column

  // ---- Stage + convert the full spike band into LDS (once) ----
  // 256 threads, 32 rows x 512 cols: 8 threads/row, 64 consecutive f32 each.
  {
    const int srow = tid >> 3;            // 0..31
    const int sk0  = (tid & 7) * 64;      // 0,64,...,448
    const float* gsrc = spk + ((size_t)(b0 + srow) * T + t) * I + sk0;
    _Float16* ldst = Ah + srow * I + sk0;
#pragma unroll
    for (int j = 0; j < 64; j += 4) {
      float4 v = *(const float4*)(gsrc + j);
      ldst[j + 0] = (_Float16)v.x;
      ldst[j + 1] = (_Float16)v.y;
      ldst[j + 2] = (_Float16)v.z;
      ldst[j + 3] = (_Float16)v.w;
    }
  }
  __syncthreads();

  v8f acc00 = {}, acc01 = {}, acc10 = {}, acc11 = {};

  for (int kk = 0; kk < I; kk += 32) {
    // A fragments (16x32 f16, ISA 7.12.2 layout) for the two M tiles
    const _Float16* ar0 = Ah + mrow * I + kk;           // rows 0..15
    const _Float16* ar1 = Ah + (mrow + 16) * I + kk;    // rows 16..31
    v8h a0lo = *(const v8h*)(ar0 + khalf);
    v8h a0hi = *(const v8h*)(ar0 + 16 + khalf);
    v8h a1lo = *(const v8h*)(ar1 + khalf);
    v8h a1hi = *(const v8h*)(ar1 + 16 + khalf);
    v16h a0, a1;
#pragma unroll
    for (int j = 0; j < 8; ++j) {
      a0[j] = a0lo[j]; a0[j + 8] = a0hi[j];
      a1[j] = a1lo[j]; a1[j + 8] = a1hi[j];
    }

    // B fragments (32x16 f16) for the wave's two N tiles; W row-major [O,I]
    const _Float16* w0 = Wh + (size_t)(n0 + (lane & 15)) * I + kk;
    const _Float16* w1 = w0 + (size_t)16 * I;
    v8h b0lo = *(const v8h*)(w0 + khalf);
    v8h b0hi = *(const v8h*)(w0 + 16 + khalf);
    v8h b1lo = *(const v8h*)(w1 + khalf);
    v8h b1hi = *(const v8h*)(w1 + 16 + khalf);
    v16h bf0, bf1;
#pragma unroll
    for (int j = 0; j < 8; ++j) {
      bf0[j] = b0lo[j]; bf0[j + 8] = b0hi[j];
      bf1[j] = b1lo[j]; bf1[j + 8] = b1hi[j];
    }

    acc00 = __builtin_amdgcn_wmma_f32_16x16x32_f16(
        false, a0, false, bf0, (short)0, acc00, false, false);
    acc01 = __builtin_amdgcn_wmma_f32_16x16x32_f16(
        false, a0, false, bf1, (short)0, acc01, false, false);
    acc10 = __builtin_amdgcn_wmma_f32_16x16x32_f16(
        false, a1, false, bf0, (short)0, acc10, false, false);
    acc11 = __builtin_amdgcn_wmma_f32_16x16x32_f16(
        false, a1, false, bf1, (short)0, acc11, false, false);
  }

  // Epilogue: C/D layout — VGPR r holds M = r + 8*(lane>=16), N = lane%16
  const int mbase = (lane >> 4) << 3;
  const int nA = n0 + (lane & 15);
  const int nB = nA + 16;
  const float biasA = bias[nA];
  const float biasB = bias[nB];
  float* outband = cur + ((size_t)t * Bsz + b0) * O;
#pragma unroll
  for (int r = 0; r < 8; ++r) {
    const int m = r + mbase;
    outband[(size_t)m * O + nA]        = acc00[r] + biasA;
    outband[(size_t)m * O + nB]        = acc01[r] + biasB;
    outband[(size_t)(m + 16) * O + nA] = acc10[r] + biasA;
    outband[(size_t)(m + 16) * O + nB] = acc11[r] + biasB;
  }
}

// ---------------------------------------------------------------------------
// Kernel 2: sequential Synaptic-LIF scan. One thread per (b,o) neuron,
// coalesced along o. snntorch step order:
//   reset = heaviside(mem_prev - th); syn = a*syn + cur;
//   mem = b*mem_prev + syn - reset*th; spk = heaviside(mem - th);
//   mem = max(mem, 0)
// ---------------------------------------------------------------------------
__global__ __launch_bounds__(256) void snn_scan(
    const float* __restrict__ cur, const float* __restrict__ alpha,
    const float* __restrict__ beta, const float* __restrict__ thr,
    float* __restrict__ spk_rec, float* __restrict__ syn_rec,
    float* __restrict__ mem_rec) {
  const int idx = blockIdx.x * 256 + threadIdx.x;  // 0 .. Bsz*O-1
  const int o = idx & (O - 1);
  const float al = alpha[o];
  const float be = beta[o];
  const float th = thr[o];

  float syn = 0.f, mem = 0.f;
  const int stride = Bsz * O;
  for (int t = 0; t < T; ++t) {
    const size_t off = (size_t)t * stride + idx;
    const float c = cur[off];
    const float reset = (mem - th > 0.f) ? 1.f : 0.f;  // from PREVIOUS mem
    syn = al * syn + c;
    mem = be * mem + syn - reset * th;
    const float s = (mem - th > 0.f) ? 1.f : 0.f;
    mem = fmaxf(mem, 0.f);
    spk_rec[off] = s;
    syn_rec[off] = syn;
    mem_rec[off] = mem;
  }
}

// ---------------------------------------------------------------------------
extern "C" void kernel_launch(void* const* d_in, const int* in_sizes, int n_in,
                              void* d_out, int out_size, void* d_ws,
                              size_t ws_size, hipStream_t stream) {
  (void)in_sizes; (void)n_in; (void)out_size; (void)ws_size;

  const float* spk   = (const float*)d_in[0];  // [B,T,I]
  const float* W     = (const float*)d_in[1];  // [O,I]
  const float* bias  = (const float*)d_in[2];  // [O]
  const float* alpha = (const float*)d_in[3];  // [O]
  const float* beta  = (const float*)d_in[4];  // [O]
  const float* thr   = (const float*)d_in[5];  // [O]

  float* out = (float*)d_out;                  // 4 records, each [T,B,O]
  float* cur_rec = out;
  float* spk_rec = out + (size_t)SZ;
  float* syn_rec = out + (size_t)2 * SZ;
  float* mem_rec = out + (size_t)3 * SZ;

  _Float16* Wh = (_Float16*)d_ws;              // 256 KB f16 weights

  snn_convert_w_f16<<<(O * I + 255) / 256, 256, 0, stream>>>(W, Wh);
  snn_cur_gemm<<<(T * Bsz) / MROWS, 256, 0, stream>>>(spk, Wh, bias, cur_rec);
  snn_scan<<<(Bsz * O) / 256, 256, 0, stream>>>(cur_rec, alpha, beta, thr,
                                                spk_rec, syn_rec, mem_rec);
}
